// GPTJAttention_39221641347299
// MI455X (gfx1250) — compile-verified, matrix operand order checked by reading, not run on hardware
//
#include <hip/hip_runtime.h>
#include <hip/hip_bf16.h>
#include <cstdint>

// ---------------------------------------------------------------------------
// Types
// ---------------------------------------------------------------------------
typedef __bf16 v16bf __attribute__((ext_vector_type(16)));
typedef float  v8f   __attribute__((ext_vector_type(8)));
typedef unsigned int u32x4 __attribute__((ext_vector_type(4)));
typedef unsigned int uv4   __attribute__((ext_vector_type(4)));
typedef int          iv8   __attribute__((ext_vector_type(8)));
typedef int          iv4   __attribute__((ext_vector_type(4)));
typedef unsigned short u16;

union FragBf { v16bf v; u32x4 q[2]; };

// Tensor Data Mover availability (this toolchain: 6-arg clang-23 form)
#if defined(__has_builtin)
#  if __has_builtin(__builtin_amdgcn_tensor_load_to_lds) && \
      __has_builtin(__builtin_amdgcn_s_wait_tensorcnt)
#    define HAVE_TDM 1
#  endif
#endif
#ifndef HAVE_TDM
#  define HAVE_TDM 0
#endif

__device__ __forceinline__ u16 f2bf(float f) {
  unsigned int u = __float_as_uint(f);
  u = (u + 0x7FFFu + ((u >> 16) & 1u)) >> 16;   // round-to-nearest-even
  return (u16)u;
}

__device__ __forceinline__ v8f wmma_bf16(const FragBf& a, const FragBf& b, v8f c) {
  return __builtin_amdgcn_wmma_f32_16x16x32_bf16(
      /*neg_a=*/false, a.v, /*neg_b=*/false, b.v,
      /*c_mod=*/(short)0, c, /*reuse_a=*/false, /*reuse_b=*/false);
}

#if HAVE_TDM
// Low 32 bits of a generic pointer to LDS == LDS byte offset (aperture form).
__device__ __forceinline__ unsigned lds_off(const void* p) {
  return (unsigned)(uintptr_t)p;
}

// 2D tile DMA: tile_d0 contiguous elements per row (2-byte elems), tile_d1 rows,
// global row stride = stride_elems. LDS destination padded by pad_amount_code
// (+1 DWORDs) after every (2<<pad_interval_code) DWORDs.
__device__ __forceinline__ void tdm_load_2d(
    unsigned lds_addr, const void* gaddr,
    unsigned tile_d0, unsigned tile_d1, unsigned stride_elems,
    unsigned pad_interval_code, unsigned pad_amount_code)
{
  const unsigned long long ga = (unsigned long long)gaddr;
  uv4 g0;
  g0[0] = 1u;                                   // count=1 (valid), user desc
  g0[1] = lds_addr;                             // lds_addr (bytes)
  g0[2] = (unsigned)(ga & 0xFFFFFFFFu);         // global_addr[31:0]
  g0[3] = (unsigned)(ga >> 32) | 0x80000000u;   // global_addr[56:32] | type=2
  iv8 g1;
  g1[0] = (int)((1u << 16)                      // data_size = 2 bytes
              | (1u << 20)                      // pad_enable
              | (pad_interval_code << 22)
              | (pad_amount_code << 25));
  g1[1] = (int)((tile_d0 & 0xFFFFu) << 16);     // tensor_dim0[15:0] @ bits 63:48
  g1[2] = (int)(((tile_d0 >> 16) & 0xFFFFu)     // tensor_dim0[31:16]
              | ((tile_d1 & 0xFFFFu) << 16));   // tensor_dim1[15:0]
  g1[3] = (int)(((tile_d1 >> 16) & 0xFFFFu)     // tensor_dim1[31:16]
              | ((tile_d0 & 0xFFFFu) << 16));   // tile_dim0
  g1[4] = (int)(tile_d1 & 0xFFFFu);             // tile_dim1 (tile_dim2 = 0)
  g1[5] = (int)stride_elems;                    // tensor_dim0_stride[31:0]
  g1[6] = 0;                                    // stride hi / dim1_stride lo
  g1[7] = 0;
  const iv4 z4 = {0, 0, 0, 0};
  const iv8 z8 = {0, 0, 0, 0, 0, 0, 0, 0};
  // 6-arg form: (g0, g1, g2, g3, extra, cpol)
  __builtin_amdgcn_tensor_load_to_lds(g0, g1, z4, z4, z8, 0);
}
#endif

// Problem constants
constexpr int B_ = 2, S_ = 2048, D_ = 4096, H_ = 16, HD_ = 256;
constexpr int N_QKV = 3 * D_;                 // 12288
constexpr float SM_SCALE = 0.0625f;           // 256^-0.5

// Workspace layout (bytes). Total = 256 MiB.
constexpr size_t OFF_X     = 0;                          // X bf16, later reused as Attn bf16
constexpr size_t OFF_WQKVT = 33554432;                   // WqkvT bf16 [12288][4096]
constexpr size_t OFF_WOUTT = OFF_WQKVT + 100663296;      // WoutT bf16 [4096][4096]
constexpr size_t OFF_Q     = OFF_WOUTT + 33554432;       // Q bf16 [B*H][S][256]
constexpr size_t OFF_K     = OFF_Q + 33554432;           // K bf16 [B*H][S][256]
constexpr size_t OFF_VT    = OFF_K + 33554432;           // Vt bf16 [B*H][256][S]

// ---------------------------------------------------------------------------
// Elementwise fp32 -> bf16 cast
// ---------------------------------------------------------------------------
__global__ void cast_bf16_kernel(const float* __restrict__ in, u16* __restrict__ out, int n) {
  int i = blockIdx.x * blockDim.x + threadIdx.x;
  int stride = gridDim.x * blockDim.x;
  for (; i < n; i += stride) out[i] = f2bf(in[i]);
}

// ---------------------------------------------------------------------------
// Cast + transpose: W[K][N] fp32 -> Wt[N][K] bf16   (32x32 tiles)
// ---------------------------------------------------------------------------
__global__ __launch_bounds__(256)
void transpose_cast_kernel(const float* __restrict__ W, u16* __restrict__ Wt, int K, int N) {
  __shared__ float tile[32][33];
  const int n0 = blockIdx.x * 32, k0 = blockIdx.y * 32;
  const int tx = threadIdx.x, ty = threadIdx.y;   // 32 x 8
  #pragma unroll
  for (int i = 0; i < 4; ++i)
    tile[ty + 8 * i][tx] = W[(size_t)(k0 + ty + 8 * i) * N + n0 + tx];
  __syncthreads();
  #pragma unroll
  for (int i = 0; i < 4; ++i)
    Wt[(size_t)(n0 + ty + 8 * i) * K + k0 + tx] = f2bf(tile[tx][ty + 8 * i]);
}

// ---------------------------------------------------------------------------
// WMMA GEMM mainloop: 128x128 block tile, 4 waves x (64x64 = 4x4 WMMAs),
// K-step 32.  A [M][K] bf16 row-major, Bt [N][K] bf16 (B pre-transposed).
// LDS rows padded to 40 u16 (80 B).  TDM path: double-buffered async tile DMA
// overlapped with WMMA; fallback: register-staged manual copies.
// sA / sB must each hold 2 * 128 * 40 u16.
// ---------------------------------------------------------------------------
__device__ __forceinline__ void wmma_mainloop(
    const u16* __restrict__ A, const u16* __restrict__ Bt, int Kdim,
    int m0, int n0, u16* sA, u16* sB, v8f acc[4][4])
{
  const int t = threadIdx.x;                 // 0..127
  const int wave = t >> 5, lane = t & 31;
  const int hl = lane >> 4, l16 = lane & 15;
  const int wm = (wave >> 1) * 64, wn = (wave & 1) * 64;

  const v8f vzero = {};
  #pragma unroll
  for (int i = 0; i < 4; ++i)
    #pragma unroll
    for (int j = 0; j < 4; ++j) acc[i][j] = vzero;

#if HAVE_TDM
  const unsigned ldsA = lds_off(sA), ldsB = lds_off(sB);
  if (wave == 0) {                                     // prologue: tile k0=0 -> buf 0
    tdm_load_2d(ldsA, A + (size_t)m0 * Kdim, 32, 128, (unsigned)Kdim, 3, 3);
    tdm_load_2d(ldsB, Bt + (size_t)n0 * Kdim, 32, 128, (unsigned)Kdim, 3, 3);
    __builtin_amdgcn_s_wait_tensorcnt(0);
  }
  __syncthreads();

  int cur = 0;
  for (int k0 = 0; k0 < Kdim; k0 += 32) {
    if (wave == 0 && (k0 + 32 < Kdim)) {               // async-prefetch next tile
      const unsigned nb = (unsigned)(cur ^ 1) * 10240u;  // buffer byte offset
      tdm_load_2d(ldsA + nb, A + (size_t)m0 * Kdim + k0 + 32, 32, 128, (unsigned)Kdim, 3, 3);
      tdm_load_2d(ldsB + nb, Bt + (size_t)n0 * Kdim + k0 + 32, 32, 128, (unsigned)Kdim, 3, 3);
    }
    const u16* cA = sA + cur * 5120;
    const u16* cB = sB + cur * 5120;

    FragBf af[4], bfr[4];
    #pragma unroll
    for (int i = 0; i < 4; ++i) {
      int ar = wm + i * 16 + l16;
      af[i].q[0] = *(const u32x4*)(cA + ar * 40 + 8 * hl);
      af[i].q[1] = *(const u32x4*)(cA + ar * 40 + 16 + 8 * hl);
      int br = wn + i * 16 + l16;
      bfr[i].q[0] = *(const u32x4*)(cB + br * 40 + 8 * hl);
      bfr[i].q[1] = *(const u32x4*)(cB + br * 40 + 16 + 8 * hl);
    }
    #pragma unroll
    for (int mt = 0; mt < 4; ++mt)
      #pragma unroll
      for (int nt = 0; nt < 4; ++nt)
        acc[mt][nt] = wmma_bf16(af[mt], bfr[nt], acc[mt][nt]);

    if (wave == 0) __builtin_amdgcn_s_wait_tensorcnt(0);  // next tile landed
    __syncthreads();
    cur ^= 1;
  }
#else
  for (int k0 = 0; k0 < Kdim; k0 += 32) {
    __syncthreads();
    u32x4 ra[4], rb[4];
    #pragma unroll
    for (int i = 0; i < 4; ++i) {            // batch global loads before stores
      int c = i * 128 + t;
      int row = c >> 2, cb = (c & 3) * 8;
      ra[i] = *(const u32x4*)(A + (size_t)(m0 + row) * Kdim + k0 + cb);
      rb[i] = *(const u32x4*)(Bt + (size_t)(n0 + row) * Kdim + k0 + cb);
    }
    #pragma unroll
    for (int i = 0; i < 4; ++i) {
      int c = i * 128 + t;
      int row = c >> 2, cb = (c & 3) * 8;
      *(u32x4*)(sA + row * 40 + cb) = ra[i];
      *(u32x4*)(sB + row * 40 + cb) = rb[i];
    }
    if (k0 + 32 < Kdim) {
      __builtin_prefetch(A + (size_t)(m0 + t) * Kdim + k0 + 32, 0, 0);
      __builtin_prefetch(Bt + (size_t)(n0 + t) * Kdim + k0 + 32, 0, 0);
    }
    __syncthreads();

    FragBf af[4], bfr[4];
    #pragma unroll
    for (int i = 0; i < 4; ++i) {
      int ar = wm + i * 16 + l16;
      af[i].q[0] = *(const u32x4*)(sA + ar * 40 + 8 * hl);
      af[i].q[1] = *(const u32x4*)(sA + ar * 40 + 16 + 8 * hl);
      int br = wn + i * 16 + l16;
      bfr[i].q[0] = *(const u32x4*)(sB + br * 40 + 8 * hl);
      bfr[i].q[1] = *(const u32x4*)(sB + br * 40 + 16 + 8 * hl);
    }
    #pragma unroll
    for (int mt = 0; mt < 4; ++mt)
      #pragma unroll
      for (int nt = 0; nt < 4; ++nt)
        acc[mt][nt] = wmma_bf16(af[mt], bfr[nt], acc[mt][nt]);
  }
#endif
}

// ---------------------------------------------------------------------------
// QKV projection GEMM + GPT-J RoPE epilogue + scatter to Q / K / Vt (bf16).
//   grid (96, 32), block 128.
// ---------------------------------------------------------------------------
__global__ __launch_bounds__(128)
void qkv_gemm_rope_kernel(const u16* __restrict__ X, const u16* __restrict__ WqkvT,
                          const int* __restrict__ posids,
                          u16* __restrict__ Qb, u16* __restrict__ Kb, u16* __restrict__ Vt)
{
  __shared__ __align__(16) u16 sA[2 * 128 * 40];
  __shared__ __align__(16) u16 sB[2 * 128 * 40];
  v8f acc[4][4];
  wmma_mainloop(X, WqkvT, D_, blockIdx.y * 128, blockIdx.x * 128, sA, sB, acc);

  const int t = threadIdx.x;
  const int wave = t >> 5, lane = t & 31;
  const int hl = lane >> 4, l16 = lane & 15;
  const int wm = (wave >> 1) * 64, wn = (wave & 1) * 64;
  const int mbase = blockIdx.y * 128 + wm;
  const int nbase = blockIdx.x * 128 + wn;

  #pragma unroll
  for (int nt = 0; nt < 4; ++nt) {
    const int col = nbase + nt * 16 + l16;
    const int part = col >> 12;              // 0=q 1=k 2=v
    const int c2 = col & 4095;
    const int h = c2 >> 8;
    const int d = c2 & 255;
    const bool rope = (part != 2) && (d < 64);   // wave-uniform
    const float invf = __expf((float)(d >> 1) * (-9.2103403719761836f / 32.0f));
    #pragma unroll
    for (int mt = 0; mt < 4; ++mt) {
      #pragma unroll
      for (int r = 0; r < 8; ++r) {
        const int m = mbase + mt * 16 + 8 * hl + r;
        float val = acc[mt][nt][r];
        if (rope) {
          float partner = __shfl_xor(val, 1);        // paired column is lane^1
          float pos = (float)posids[m];
          float ang = pos * invf;
          float sn, cs;
          __sincosf(ang, &sn, &cs);
          val = (d & 1) ? (val * cs + partner * sn) : (val * cs - partner * sn);
        }
        const u16 bv = f2bf(val);
        const int b = m >> 11, s = m & 2047;
        const int bh = (b << 4) + h;
        if (part == 0)      Qb[((size_t)bh * S_ + s) * HD_ + d] = bv;
        else if (part == 1) Kb[((size_t)bh * S_ + s) * HD_ + d] = bv;
        else                Vt[((size_t)bh * HD_ + d) * S_ + s] = bv;
      }
    }
  }
}

// ---------------------------------------------------------------------------
// Flash attention.  grid (S/32, B*H), block 64 (2 waves).
// Each wave owns 16 q rows; loop over 32-key tiles with online softmax.
// LDS: sQ 32x264, sK 32x264, sV 256x40, sP 2x512  => 56320 B.
// ---------------------------------------------------------------------------
__global__ __launch_bounds__(64)
void flash_attn_kernel(const u16* __restrict__ Qb, const u16* __restrict__ Kb,
                       const u16* __restrict__ Vt, u16* __restrict__ Attn)
{
  __shared__ __align__(16) u16 sQ[32 * 264];
  __shared__ __align__(16) u16 sK[32 * 264];
  __shared__ __align__(16) u16 sV[256 * 40];
  __shared__ __align__(16) u16 sP[2 * 512];

  const int t = threadIdx.x;                 // 0..63
  const int wq = t >> 5, lane = t & 31;
  const int hl = lane >> 4, l16 = lane & 15;
  const int bh = blockIdx.y;                 // b*16 + h
  const int q0 = blockIdx.x * 32;

  // load Q tile (32 rows x 256) once
#if HAVE_TDM
  if (wq == 0) {
    tdm_load_2d(lds_off(sQ), Qb + ((size_t)bh * S_ + q0) * HD_, 256, 32, 256, 6, 3);
    __builtin_amdgcn_s_wait_tensorcnt(0);
  }
#else
  #pragma unroll
  for (int i = 0; i < 16; ++i) {
    int c = i * 64 + t;                      // 1024 chunks of 8 u16
    int row = c >> 5, cb = (c & 31) * 8;
    *(u32x4*)(sQ + row * 264 + cb) =
        *(const u32x4*)(Qb + ((size_t)bh * S_ + q0 + row) * HD_ + cb);
  }
#endif

  v8f o[16];
  const v8f vzero = {};
  #pragma unroll
  for (int i = 0; i < 16; ++i) o[i] = vzero;
  float mrun[8], lrun[8];
  #pragma unroll
  for (int r = 0; r < 8; ++r) { mrun[r] = -1e30f; lrun[r] = 0.0f; }

  const int kend = q0 + 32;                  // causal: keys <= q0+31
  for (int kt0 = 0; kt0 < kend; kt0 += 32) {
    __syncthreads();                         // previous iteration's reads done
#if HAVE_TDM
    if (wq == 0) {
      tdm_load_2d(lds_off(sK), Kb + ((size_t)bh * S_ + kt0) * HD_, 256, 32, 256, 6, 3);
      tdm_load_2d(lds_off(sV), Vt + (size_t)bh * HD_ * S_ + kt0, 32, 256, 2048, 3, 3);
      __builtin_amdgcn_s_wait_tensorcnt(0);
    }
#else
    #pragma unroll
    for (int i = 0; i < 16; ++i) {
      int c = i * 64 + t;
      { int row = c >> 5, cb = (c & 31) * 8;                     // K tile [32][256]
        *(u32x4*)(sK + row * 264 + cb) =
            *(const u32x4*)(Kb + ((size_t)bh * S_ + kt0 + row) * HD_ + cb); }
      { int d = c >> 2, cb = (c & 3) * 8;                        // V tile [256][32]
        *(u32x4*)(sV + d * 40 + cb) =
            *(const u32x4*)(Vt + ((size_t)bh * HD_ + d) * S_ + kt0 + cb); }
    }
#endif
    __syncthreads();

    // scores: 16 q rows x 32 keys
    v8f sc0 = vzero, sc1 = vzero;
    const int qrow = wq * 16 + l16;
    #pragma unroll
    for (int ks = 0; ks < 8; ++ks) {
      FragBf qa, kb0, kb1;
      qa.q[0]  = *(const u32x4*)(sQ + qrow * 264 + ks * 32 + 8 * hl);
      qa.q[1]  = *(const u32x4*)(sQ + qrow * 264 + ks * 32 + 16 + 8 * hl);
      kb0.q[0] = *(const u32x4*)(sK + l16 * 264 + ks * 32 + 8 * hl);
      kb0.q[1] = *(const u32x4*)(sK + l16 * 264 + ks * 32 + 16 + 8 * hl);
      kb1.q[0] = *(const u32x4*)(sK + (16 + l16) * 264 + ks * 32 + 8 * hl);
      kb1.q[1] = *(const u32x4*)(sK + (16 + l16) * 264 + ks * 32 + 16 + 8 * hl);
      sc0 = wmma_bf16(qa, kb0, sc0);
      sc1 = wmma_bf16(qa, kb1, sc1);
    }

    // online softmax (per-row over the 16-lane column groups)
    float ps0[8], ps1[8], rm[8];
    #pragma unroll
    for (int r = 0; r < 8; ++r) {
      const int qi = q0 + wq * 16 + 8 * hl + r;
      const int j0 = kt0 + l16, j1 = kt0 + 16 + l16;
      ps0[r] = (j0 <= qi) ? sc0[r] * SM_SCALE : -1e30f;
      ps1[r] = (j1 <= qi) ? sc1[r] * SM_SCALE : -1e30f;
      float mx = fmaxf(ps0[r], ps1[r]);
      mx = fmaxf(mx, __shfl_xor(mx, 1));
      mx = fmaxf(mx, __shfl_xor(mx, 2));
      mx = fmaxf(mx, __shfl_xor(mx, 4));
      mx = fmaxf(mx, __shfl_xor(mx, 8));
      rm[r] = mx;
    }
    #pragma unroll
    for (int r = 0; r < 8; ++r) {
      const float mn = fmaxf(mrun[r], rm[r]);
      const float alpha = __expf(mrun[r] - mn);
      mrun[r] = mn;
      lrun[r] *= alpha;
      #pragma unroll
      for (int dblk = 0; dblk < 16; ++dblk) o[dblk][r] *= alpha;
      const float p0 = __expf(ps0[r] - mn);
      const float p1 = __expf(ps1[r] - mn);
      ps0[r] = p0; ps1[r] = p1;
      float rs = p0 + p1;
      rs += __shfl_xor(rs, 1);
      rs += __shfl_xor(rs, 2);
      rs += __shfl_xor(rs, 4);
      rs += __shfl_xor(rs, 8);
      lrun[r] += rs;
    }

    // C-layout -> A-layout for P via LDS (wave-private region)
    #pragma unroll
    for (int r = 0; r < 8; ++r) {
      const int rl = 8 * hl + r;
      sP[wq * 512 + rl * 32 + l16]      = f2bf(ps0[r]);
      sP[wq * 512 + rl * 32 + 16 + l16] = f2bf(ps1[r]);
    }
    FragBf pf;
    pf.q[0] = *(const u32x4*)(sP + wq * 512 + l16 * 32 + 8 * hl);
    pf.q[1] = *(const u32x4*)(sP + wq * 512 + l16 * 32 + 16 + 8 * hl);

    // O += P (16x32) @ V (32x256)
    #pragma unroll
    for (int dblk = 0; dblk < 16; ++dblk) {
      FragBf vf;
      vf.q[0] = *(const u32x4*)(sV + (dblk * 16 + l16) * 40 + 8 * hl);
      vf.q[1] = *(const u32x4*)(sV + (dblk * 16 + l16) * 40 + 16 + 8 * hl);
      o[dblk] = wmma_bf16(pf, vf, o[dblk]);
    }
  }

  // normalize + store attn [B,S,H*hd] as bf16
  float invl[8];
  #pragma unroll
  for (int r = 0; r < 8; ++r) invl[r] = 1.0f / lrun[r];
  const int b = bh >> 4, h = bh & 15;
  #pragma unroll
  for (int dblk = 0; dblk < 16; ++dblk) {
    #pragma unroll
    for (int r = 0; r < 8; ++r) {
      const int qi = q0 + wq * 16 + 8 * hl + r;
      Attn[((size_t)(b * S_ + qi)) * D_ + h * HD_ + dblk * 16 + l16] =
          f2bf(o[dblk][r] * invl[r]);
    }
  }
}

// ---------------------------------------------------------------------------
// Output projection: Attn [4096][4096] bf16 @ w_out -> fp32 d_out.
//   grid (32, 32), block 128.
// ---------------------------------------------------------------------------
__global__ __launch_bounds__(128)
void out_gemm_kernel(const u16* __restrict__ Attn, const u16* __restrict__ WoutT,
                     float* __restrict__ out)
{
  __shared__ __align__(16) u16 sA[2 * 128 * 40];
  __shared__ __align__(16) u16 sB[2 * 128 * 40];
  v8f acc[4][4];
  wmma_mainloop(Attn, WoutT, D_, blockIdx.y * 128, blockIdx.x * 128, sA, sB, acc);

  const int t = threadIdx.x;
  const int wave = t >> 5, lane = t & 31;
  const int hl = lane >> 4, l16 = lane & 15;
  const int wm = (wave >> 1) * 64, wn = (wave & 1) * 64;
  const int mbase = blockIdx.y * 128 + wm;
  const int nbase = blockIdx.x * 128 + wn;

  #pragma unroll
  for (int mt = 0; mt < 4; ++mt)
    #pragma unroll
    for (int nt = 0; nt < 4; ++nt)
      #pragma unroll
      for (int r = 0; r < 8; ++r) {
        const int m = mbase + mt * 16 + 8 * hl + r;
        const int n = nbase + nt * 16 + l16;
        out[(size_t)m * D_ + n] = acc[mt][nt][r];
      }
}

// ---------------------------------------------------------------------------
// Host launcher
// ---------------------------------------------------------------------------
extern "C" void kernel_launch(void* const* d_in, const int* in_sizes, int n_in,
                              void* d_out, int out_size, void* d_ws, size_t ws_size,
                              hipStream_t stream) {
  const int*   posids = (const int*)d_in[0];
  const float* hidden = (const float*)d_in[1];
  const float* wqkv   = (const float*)d_in[2];
  const float* wout   = (const float*)d_in[3];
  float* out = (float*)d_out;

  char* ws = (char*)d_ws;
  u16* X     = (u16*)(ws + OFF_X);
  u16* WqkvT = (u16*)(ws + OFF_WQKVT);
  u16* WoutT = (u16*)(ws + OFF_WOUTT);
  u16* Qb    = (u16*)(ws + OFF_Q);
  u16* Kb    = (u16*)(ws + OFF_K);
  u16* Vt    = (u16*)(ws + OFF_VT);
  u16* Attn  = (u16*)(ws + OFF_X);   // reuse X slot after QKV GEMM consumes it

  // 1. casts / transposes
  cast_bf16_kernel<<<2048, 256, 0, stream>>>(hidden, X, B_ * S_ * D_);
  transpose_cast_kernel<<<dim3(N_QKV / 32, D_ / 32), dim3(32, 8), 0, stream>>>(
      wqkv, WqkvT, D_, N_QKV);
  transpose_cast_kernel<<<dim3(D_ / 32, D_ / 32), dim3(32, 8), 0, stream>>>(
      wout, WoutT, D_, D_);

  // 2. QKV projection + RoPE
  qkv_gemm_rope_kernel<<<dim3(N_QKV / 128, (B_ * S_) / 128), 128, 0, stream>>>(
      X, WqkvT, posids, Qb, Kb, Vt);

  // 3. flash attention
  flash_attn_kernel<<<dim3(S_ / 32, B_ * H_), 64, 0, stream>>>(Qb, Kb, Vt, Attn);

  // 4. output projection
  out_gemm_kernel<<<dim3(D_ / 128, (B_ * S_) / 128), 128, 0, stream>>>(Attn, WoutT, out);
}
